// MultiHeadAttention_74826920230963
// MI455X (gfx1250) — matrix operand order repaired and checked by reading, not compiled, not run on hardware
//
#include <hip/hip_runtime.h>

typedef __bf16 bf16;
typedef __attribute__((ext_vector_type(8)))  __bf16 v8bf;
typedef __attribute__((ext_vector_type(16))) __bf16 v16bf;
typedef __attribute__((ext_vector_type(8)))  float  v8f;

#define DEV __device__ __forceinline__

static constexpr int Bb  = 2;
static constexpr int Ss  = 2048;
static constexpr int DMm = 1024;
static constexpr int Hh  = 16;
// DH = 64

// ---- float -> bf16 (round to nearest even) ----
DEV bf16 f2bf(float f) {
  unsigned u = __builtin_bit_cast(unsigned, f);
  unsigned r = u + 0x7FFFu + ((u >> 16) & 1u);
  unsigned short h = (unsigned short)(r >> 16);
  return __builtin_bit_cast(bf16, h);
}

// ---- WMMA fragment loaders matching CDNA5 ISA VGPR layouts (wave32) ----
// A fragment: 16x32 (MxK), 16-bit. Lane L<16: row M=L, K={k0..k0+7, k0+16..k0+23}.
// Lane L>=16: row M=L-16, K={k0+8..k0+15, k0+24..k0+31}.
DEV v16bf load_frag_a(const bf16* A, int ld, int row0, int k0, int lane) {
  const int r  = row0 + (lane & 15);
  const int kk = k0 + ((lane >> 4) << 3);
  const bf16* p = A + (size_t)r * ld + kk;
  v8bf lo = *(const v8bf*)(p);
  v8bf hi = *(const v8bf*)(p + 16);
  return __builtin_shufflevector(lo, hi, 0,1,2,3,4,5,6,7,8,9,10,11,12,13,14,15);
}

// B fragment: 32x16 (KxN), 16-bit, from a K-major matrix Wm[n][k] (B[k][n]=Wm[n][k]).
// Lane L<16: col N=n0+L, K={k0..k0+7, k0+16..k0+23}. Lane L>=16: col N=n0+L-16, K={k0+8..k0+15, k0+24..k0+31}.
DEV v16bf load_frag_b(const bf16* Wm, int ld, int n0, int k0, int lane) {
  const int n  = n0 + (lane & 15);
  const int kk = k0 + ((lane >> 4) << 3);
  const bf16* p = Wm + (size_t)n * ld + kk;
  v8bf lo = *(const v8bf*)(p);
  v8bf hi = *(const v8bf*)(p + 16);
  return __builtin_shufflevector(lo, hi, 0,1,2,3,4,5,6,7,8,9,10,11,12,13,14,15);
}

DEV v8f wmma_bf16(v16bf a, v16bf b, v8f c) {
  return __builtin_amdgcn_wmma_f32_16x16x32_bf16(false, a, false, b, (short)0, c, false, false);
}

// ---- CDNA5 async global->LDS (ASYNCcnt-tracked), 16B per lane ----
DEV void async_copy_b128(const bf16* lds_dst, const bf16* gsrc) {
  unsigned loff = (unsigned)(size_t)lds_dst;   // low 32 bits of generic ptr = LDS offset
  asm volatile("global_load_async_to_lds_b128 %0, %1, off"
               :: "v"(loff), "v"(gsrc)
               : "memory");
}

DEV void wait_async0() {
#if __has_builtin(__builtin_amdgcn_s_wait_asynccnt)
  __builtin_amdgcn_s_wait_asynccnt(0);
#else
  asm volatile("s_wait_asynccnt 0x0" ::: "memory");
#endif
}

// ---- fp32 -> bf16 conversion kernel ----
__global__ __launch_bounds__(256) void cvt_f32_bf16(const float* __restrict__ in,
                                                    bf16* __restrict__ out, int n) {
  int i = blockIdx.x * blockDim.x + threadIdx.x;
  const int stride = gridDim.x * blockDim.x;
  for (; i < n; i += stride) out[i] = f2bf(in[i]);
}

// ---- GEMM: out[m][n] = sum_k A[m][k]*Wm[n][k] + bias[n] ----
// Block tile 128x64 (8 waves, each 32x32), K-step 32, double-buffered LDS
// filled with global_load_async_to_lds_b128.
// mode 1: bf16 out -> q/k layout [b][h][s][d]
// mode 2: bf16 out -> vT layout  [b][h][d][s]
// mode 3: f32 out, row-major [M][N]
__global__ __launch_bounds__(256) void gemm_bf16_wmma(
    const bf16* __restrict__ A, const bf16* __restrict__ Wm,
    const float* __restrict__ bias, void* __restrict__ out,
    int M, int N, int K, int mode)
{
  __shared__ bf16 Asmem[2][128][32];   // 16 KB
  __shared__ bf16 Bsmem[2][64][32];    //  8 KB

  const int tid  = threadIdx.x;
  const int lane = tid & 31;
  const int wave = tid >> 5;
  const int wm = wave & 3;             // 4 waves along M
  const int wn = wave >> 2;            // 2 waves along N
  const int rowB = blockIdx.x * 128;
  const int colB = blockIdx.y * 64;

  // per-thread async-fill coordinates (16B chunks, 4 chunks per 32-elem row)
  const int ar0 = tid >> 2;                  // A rows 0..63
  const int ar1 = (tid + 256) >> 2;          // A rows 64..127
  const int ac  = (tid & 3) * 8;
  const int br  = tid >> 2;                  // B rows 0..63

  const bf16* gA0 = A  + (size_t)(rowB + ar0) * K + ac;
  const bf16* gA1 = A  + (size_t)(rowB + ar1) * K + ac;
  const bf16* gB  = Wm + (size_t)(colB + br ) * K + ac;

  // prologue: fill buffer 0 with k0 = 0
  async_copy_b128(&Asmem[0][ar0][ac], gA0);
  async_copy_b128(&Asmem[0][ar1][ac], gA1);
  async_copy_b128(&Bsmem[0][br ][ac], gB);

  v8f acc00 = {}, acc01 = {}, acc10 = {}, acc11 = {};
  int buf = 0;
  for (int k0 = 0; k0 < K; k0 += 32, buf ^= 1) {
    wait_async0();        // my fills of Asmem/Bsmem[buf] done
    __syncthreads();      // everyone's fills done; everyone done reading buf^1
    if (k0 + 32 < K) {    // overlap next slab's fill with this slab's WMMAs
      const int kn = k0 + 32;
      async_copy_b128(&Asmem[buf ^ 1][ar0][ac], gA0 + kn);
      async_copy_b128(&Asmem[buf ^ 1][ar1][ac], gA1 + kn);
      async_copy_b128(&Bsmem[buf ^ 1][br ][ac], gB  + kn);
    }
    const bf16* As = &Asmem[buf][0][0];
    const bf16* Bs = &Bsmem[buf][0][0];
    v16bf a0 = load_frag_a(As, 32, wm * 32,      0, lane);
    v16bf a1 = load_frag_a(As, 32, wm * 32 + 16, 0, lane);
    v16bf b0 = load_frag_b(Bs, 32, wn * 32,      0, lane);
    v16bf b1 = load_frag_b(Bs, 32, wn * 32 + 16, 0, lane);
    acc00 = wmma_bf16(a0, b0, acc00);
    acc01 = wmma_bf16(a0, b1, acc01);
    acc10 = wmma_bf16(a1, b0, acc10);
    acc11 = wmma_bf16(a1, b1, acc11);
  }

  // C layout: lanes 0-15 hold rows 0-7 (VGPR r = row r), lanes 16-31 rows 8-15.
  const int row0 = rowB + wm * 32;
  const int col0 = colB + wn * 32;
  const int rb = ((lane >> 4) << 3);
  const int cl = lane & 15;
  #pragma unroll
  for (int i = 0; i < 2; ++i) {
    #pragma unroll
    for (int j = 0; j < 2; ++j) {
      v8f accv = (i == 0) ? (j == 0 ? acc00 : acc01)
                          : (j == 0 ? acc10 : acc11);
      const int n = col0 + 16 * j + cl;
      const float bn = bias[n];
      #pragma unroll
      for (int r = 0; r < 8; ++r) {
        const int m = row0 + 16 * i + rb + r;
        const float vl = accv[r] + bn;
        if (mode == 3) {
          ((float*)out)[(size_t)m * N + n] = vl;
        } else {
          const int bI = m / Ss, sI = m % Ss;
          const int hI = n >> 6, dI = n & 63;
          const bf16 ov = f2bf(vl);
          if (mode == 1)
            ((bf16*)out)[(((size_t)bI * Hh + hI) * Ss + sI) * 64 + dI] = ov;
          else
            ((bf16*)out)[(((size_t)bI * Hh + hI) * 64 + dI) * Ss + sI] = ov;
        }
      }
    }
  }
}

// ---- causal flash attention: one 16-row query tile per wave ----
// q,k: [b*h][s][64] bf16 ; vt: [b*h][64][s] bf16 ; ctx out: [b][s][DM] bf16
__global__ __launch_bounds__(256) void flash_attn_wmma(
    const bf16* __restrict__ q, const bf16* __restrict__ k,
    const bf16* __restrict__ vt, bf16* __restrict__ ctx)
{
  __shared__ bf16 pbuf[8][16 * 32];   // per-wave P tile (16 rows x 32 t)
  const int lane = threadIdx.x & 31;
  const int wave = threadIdx.x >> 5;
  const int bh = blockIdx.y;
  const int bI = bh >> 4, hI = bh & 15;
  const int s0 = blockIdx.x * 128 + wave * 16;

  const bf16* qh = q  + (size_t)bh * Ss * 64;
  const bf16* kh = k  + (size_t)bh * Ss * 64;
  const bf16* vh = vt + (size_t)bh * 64 * Ss;

  const v16bf qf0 = load_frag_a(qh, 64, s0, 0,  lane);
  const v16bf qf1 = load_frag_a(qh, 64, s0, 32, lane);

  v8f acc0 = {}, acc1 = {}, acc2 = {}, acc3 = {};
  float mrow[8], lrow[8];
  #pragma unroll
  for (int r = 0; r < 8; ++r) { mrow[r] = -3.0e38f; lrow[r] = 0.0f; }

  const int rb = ((lane >> 4) << 3);
  const int cl = lane & 15;
  bf16* myp = &pbuf[wave][0];

  for (int t0 = 0; t0 < s0 + 16; t0 += 32) {
    // scores for t0..t0+15 (sA) and t0+16..t0+31 (sB), K-dim = DH = 64
    v8f sA = {}, sB = {};
    sA = wmma_bf16(qf0, load_frag_b(kh, 64, t0,      0,  lane), sA);
    sA = wmma_bf16(qf1, load_frag_b(kh, 64, t0,      32, lane), sA);
    sB = wmma_bf16(qf0, load_frag_b(kh, 64, t0 + 16, 0,  lane), sB);
    sB = wmma_bf16(qf1, load_frag_b(kh, 64, t0 + 16, 32, lane), sB);

    #pragma unroll
    for (int r = 0; r < 8; ++r) {
      const int srow = s0 + rb + r;
      float e0 = sA[r] * 0.125f;          // 1/sqrt(64)
      float e1 = sB[r] * 0.125f;
      if (t0 + cl > srow)      e0 = -3.0e38f;   // causal mask
      if (t0 + 16 + cl > srow) e1 = -3.0e38f;
      float mt = fmaxf(e0, e1);
      #pragma unroll
      for (int msk = 1; msk < 16; msk <<= 1)
        mt = fmaxf(mt, __shfl_xor(mt, msk, 32));  // stays within half-wave
      const float mnew = fmaxf(mrow[r], mt);
      const float fac = __expf(mrow[r] - mnew);
      const float p0 = __expf(e0 - mnew);
      const float p1 = __expf(e1 - mnew);
      float rs = p0 + p1;
      #pragma unroll
      for (int msk = 1; msk < 16; msk <<= 1)
        rs += __shfl_xor(rs, msk, 32);
      lrow[r] = lrow[r] * fac + rs;
      mrow[r] = mnew;
      acc0[r] *= fac; acc1[r] *= fac; acc2[r] *= fac; acc3[r] *= fac;
      myp[(rb + r) * 32 + cl]      = f2bf(p0);
      myp[(rb + r) * 32 + 16 + cl] = f2bf(p1);
    }

    // P (16x32) x Vt -> ctx, per-wave LDS staging (same-wave dep; HW DScnt waits)
    const v16bf pf = load_frag_a(myp, 32, 0, 0, lane);
    acc0 = wmma_bf16(pf, load_frag_b(vh, Ss, 0,  t0, lane), acc0);
    acc1 = wmma_bf16(pf, load_frag_b(vh, Ss, 16, t0, lane), acc1);
    acc2 = wmma_bf16(pf, load_frag_b(vh, Ss, 32, t0, lane), acc2);
    acc3 = wmma_bf16(pf, load_frag_b(vh, Ss, 48, t0, lane), acc3);
  }

  #pragma unroll
  for (int r = 0; r < 8; ++r) {
    const float inv = 1.0f / lrow[r];
    const int sI = s0 + rb + r;
    bf16* orow = ctx + ((size_t)bI * Ss + sI) * DMm + hI * 64;
    orow[0  + cl] = f2bf(acc0[r] * inv);
    orow[16 + cl] = f2bf(acc1[r] * inv);
    orow[32 + cl] = f2bf(acc2[r] * inv);
    orow[48 + cl] = f2bf(acc3[r] * inv);
  }
}

extern "C" void kernel_launch(void* const* d_in, const int* in_sizes, int n_in,
                              void* d_out, int out_size, void* d_ws, size_t ws_size,
                              hipStream_t stream) {
  (void)in_sizes; (void)n_in; (void)out_size; (void)ws_size;
  const float* x  = (const float*)d_in[0];
  const float* Wq = (const float*)d_in[1];
  const float* bq = (const float*)d_in[2];
  const float* Wk = (const float*)d_in[3];
  const float* bk = (const float*)d_in[4];
  const float* Wv = (const float*)d_in[5];
  const float* bv = (const float*)d_in[6];
  const float* Wo = (const float*)d_in[7];
  const float* bo = (const float*)d_in[8];
  // d_in[9] = attention_mask: pure causal, applied analytically in-kernel.
  float* out = (float*)d_out;

  const size_t NX = (size_t)Bb * Ss * DMm;  // 4,194,304
  const size_t NW = (size_t)DMm * DMm;      // 1,048,576

  bf16* xb  = (bf16*)d_ws;
  bf16* wqb = xb  + NX;
  bf16* wkb = wqb + NW;
  bf16* wvb = wkb + NW;
  bf16* wob = wvb + NW;
  bf16* qb  = wob + NW;     // [b][h][s][64]
  bf16* kb  = qb  + NX;     // [b][h][t][64]
  bf16* vtb = kb  + NX;     // [b][h][64][t]
  bf16* ctb = vtb + NX;     // [b][s][DM]
  // total ws usage: 48 MB

  const dim3 blk(256);
  cvt_f32_bf16<<<2048, blk, 0, stream>>>(x,  xb,  (int)NX);
  cvt_f32_bf16<<<1024, blk, 0, stream>>>(Wq, wqb, (int)NW);
  cvt_f32_bf16<<<1024, blk, 0, stream>>>(Wk, wkb, (int)NW);
  cvt_f32_bf16<<<1024, blk, 0, stream>>>(Wv, wvb, (int)NW);
  cvt_f32_bf16<<<1024, blk, 0, stream>>>(Wo, wob, (int)NW);

  const dim3 gg((Bb * Ss) / 128, DMm / 64);  // (32, 16)
  gemm_bf16_wmma<<<gg, blk, 0, stream>>>(xb, wqb, bq, qb,  Bb * Ss, DMm, DMm, 1);
  gemm_bf16_wmma<<<gg, blk, 0, stream>>>(xb, wkb, bk, kb,  Bb * Ss, DMm, DMm, 1);
  gemm_bf16_wmma<<<gg, blk, 0, stream>>>(xb, wvb, bv, vtb, Bb * Ss, DMm, DMm, 2);

  flash_attn_wmma<<<dim3(Ss / 128, Bb * Hh), blk, 0, stream>>>(qb, kb, vtb, ctb);

  gemm_bf16_wmma<<<gg, blk, 0, stream>>>(ctb, wob, bo, out, Bb * Ss, DMm, DMm, 3);
}
